// EdgeLLMAttention_69715909149346
// MI455X (gfx1250) — compile-verified, matrix operand order checked by reading, not run on hardware
//
#include <hip/hip_runtime.h>
#include <hip/hip_bf16.h>

typedef __bf16 bf16_t;
typedef __attribute__((ext_vector_type(16))) __bf16 bf16x16;
typedef __attribute__((ext_vector_type(8)))  float  f32x8;
typedef __attribute__((ext_vector_type(4)))  unsigned u32x4;
typedef __attribute__((ext_vector_type(8)))  int      i32x8;
typedef __attribute__((ext_vector_type(4)))  int      i32x4;

#define B_DIM  2
#define S_DIM  2048
#define HID_   4096
#define NH     32
#define NKVH   8
#define HD     128
#define MAXPOS 4096

#if defined(__has_builtin)
#if __has_builtin(__builtin_amdgcn_tensor_load_to_lds)
#define HAVE_TDM 1
#endif
#if __has_builtin(__builtin_amdgcn_s_wait_tensorcnt)
#define HAVE_TENSORCNT_BUILTIN 1
#endif
#endif

__device__ __forceinline__ void wait_tensorcnt0() {
#if defined(HAVE_TENSORCNT_BUILTIN)
  __builtin_amdgcn_s_wait_tensorcnt(0);
#else
  asm volatile("s_wait_tensorcnt 0x0" ::: "memory");
#endif
}

union FragB16 { bf16x16 v; unsigned u[8]; };

__device__ __forceinline__ f32x8 wmma_bf16(FragB16 a, FragB16 b, f32x8 c) {
  return __builtin_amdgcn_wmma_f32_16x16x32_bf16(false, a.v, false, b.v,
                                                 (short)0, c, false, false);
}

// K-index map for 16-bit fragments (ISA 7.12.2): lanes 0-15 hold K {0..7,16..23},
// lanes 16-31 hold K {8..15,24..31}; pair j packs (k, k+1) in (lo16, hi16).
__device__ __forceinline__ int frag_k(int j, int kh) {
  return (j < 4) ? (kh + 2 * j) : (16 + kh + 2 * (j - 4));
}

#if defined(HAVE_TDM)
// Issue a 2D TDM tile load (bf16 elements) into LDS with row padding.
// D# layout per CDNA5 ISA:
//   group0: [1:0]=count, [63:32]=lds_addr, [120:64]=global_addr, [127:126]=type(2)
//   group1: d0 = {pad_amount[31:25], pad_interval[24:22], pad_en[20],
//                 data_size[17:16]=1(2B), wg_mask[15:0]=0}
//           tensor_dim0[79:48], tensor_dim1[111:80], tile_dim0[127:112],
//           tile_dim1[143:128], tile_dim2[159:144]=0,
//           tensor_dim0_stride[207:160], tensor_dim1_stride[255:208]=0
// pad_interval code: 0=2,1=4,2=8,3=16,4=32,5=64,... DWORDs; pad_amount code: n -> n+1 DWORDs.
// This toolchain (clang-23 / therock-10.0) uses the 6-arg builtin:
//   (v4u g0, v8i g1, v4i g2, v4i g3, v8i extra, i32 cpol)
__device__ __forceinline__ void tdm_load_2d(unsigned lds_addr, const void* gptr,
                                            unsigned tile_x, unsigned tile_y,
                                            unsigned tdim0, unsigned tdim1,
                                            unsigned long long stride_elems,
                                            unsigned pad_interval, unsigned pad_amount) {
  unsigned long long ga = (unsigned long long)(size_t)gptr;
  u32x4 g0;
  g0.x = 1u;                                        // count=1, user mode
  g0.y = lds_addr;                                  // LDS byte address
  g0.z = (unsigned)(ga & 0xffffffffu);              // global_addr[31:0]
  g0.w = (unsigned)((ga >> 32) & 0x01ffffffu) | (2u << 30);  // addr[56:32] | type=2
  i32x8 g1;
  g1[0] = (int)((1u << 16) | (1u << 20) | (pad_interval << 22) | (pad_amount << 25));
  g1[1] = (int)((tdim0 & 0xffffu) << 16);                               // tensor_dim0 lo16
  g1[2] = (int)(((tdim0 >> 16) & 0xffffu) | ((tdim1 & 0xffffu) << 16)); // tdim0 hi / tdim1 lo
  g1[3] = (int)(((tdim1 >> 16) & 0xffffu) | ((tile_x & 0xffffu) << 16));// tdim1 hi / tile_dim0
  g1[4] = (int)(tile_y & 0xffffu);                                      // tile_dim1, tile_dim2=0
  g1[5] = (int)(unsigned)(stride_elems & 0xffffffffu);                  // dim0_stride[31:0]
  g1[6] = (int)(unsigned)((stride_elems >> 32) & 0xffffu);              // dim0_stride[47:32]
  g1[7] = 0;
  i32x4 gz = {0, 0, 0, 0};                          // groups 2/3: tile_dim3/4 = 0 (unused)
  i32x8 gz8 = {0, 0, 0, 0, 0, 0, 0, 0};             // extra group words (unused, count=1)
  __builtin_amdgcn_tensor_load_to_lds(g0, g1, gz, gz, gz8, 0);
}
#endif

// ---------------------------------------------------------------- utilities
__global__ void copy_f32x4(const float* __restrict__ src, float* __restrict__ dst, int n4) {
  int i = blockIdx.x * blockDim.x + threadIdx.x;
  if (i < n4) ((float4*)dst)[i] = ((const float4*)src)[i];
}

__global__ void cvt_f32_bf16(const float* __restrict__ src, bf16_t* __restrict__ dst, int n4) {
  int i = blockIdx.x * blockDim.x + threadIdx.x;
  if (i >= n4) return;
  float4 f = ((const float4*)src)[i];
  union { bf16_t h[4]; uint2 u; } o;
  o.h[0] = (bf16_t)f.x; o.h[1] = (bf16_t)f.y;
  o.h[2] = (bf16_t)f.z; o.h[3] = (bf16_t)f.w;
  ((uint2*)dst)[i] = o.u;
}

// ------------------------------------------------------------------- GEMM
// C[M,N] (f32) = A[M,K] (bf16, row-major) @ B[K,N] (bf16, row-major)
// block tile 128x128, 8 waves, each wave 64x32 (4x2 wmma tiles), k-step 32.
// Tile staging via the Tensor Data Mover when available (TENSORcnt path).
__launch_bounds__(256)
__global__ void gemm_bf16_wmma(const bf16_t* __restrict__ A,
                               const bf16_t* __restrict__ Bm,
                               float* __restrict__ C, int M, int N, int K) {
  __shared__ bf16_t As[128 * 40];   // [row][k] stride 40 halves (32 + 8 pad)
  __shared__ bf16_t Bs[32 * 136];   // [k][n]   stride 136 halves (128 + 8 pad)
  const int tid  = threadIdx.x;
  const int lane = tid & 31;
  const int w    = tid >> 5;
  const int mw   = w >> 2, nw = w & 3;
  const int m0   = blockIdx.y * 128, n0 = blockIdx.x * 128;
  const int r    = lane & 15, kh = (lane >> 4) * 8;

  f32x8 acc[4][2] = {};

  for (int k0 = 0; k0 < K; k0 += 32) {
    __syncthreads();
#if defined(HAVE_TDM)
    if (w == 0) {
      // A tile: 32 x 128 rows, 16-DW rows + 4-DW pad -> stride 40 halves
      tdm_load_2d((unsigned)(size_t)As, A + (size_t)m0 * K + k0,
                  /*tile_x=*/32, /*tile_y=*/128,
                  (unsigned)(K - k0), (unsigned)(M - m0),
                  (unsigned long long)K, /*pad_interval=*/3, /*pad_amount=*/3);
      // B tile: 128 x 32 rows, 64-DW rows + 4-DW pad -> stride 136 halves
      tdm_load_2d((unsigned)(size_t)Bs, Bm + (size_t)k0 * N + n0,
                  /*tile_x=*/128, /*tile_y=*/32,
                  (unsigned)(N - n0), (unsigned)(K - k0),
                  (unsigned long long)N, /*pad_interval=*/5, /*pad_amount=*/3);
      wait_tensorcnt0();
    }
#else
    // stage A tile 128x32 (512 dword4)
    for (int v = tid; v < 512; v += 256) {
      int row = v >> 2, kq = (v & 3) * 8;
      uint4 d = *(const uint4*)(A + (size_t)(m0 + row) * K + k0 + kq);
      *(uint4*)(As + row * 40 + kq) = d;
      if (k0 + 32 < K)
        __builtin_prefetch(A + (size_t)(m0 + row) * K + k0 + 32 + kq, 0, 1);
    }
    // stage B tile 32x128 (512 dword4)
    for (int v = tid; v < 512; v += 256) {
      int krow = v >> 4, nq = (v & 15) * 8;
      uint4 d = *(const uint4*)(Bm + (size_t)(k0 + krow) * N + n0 + nq);
      *(uint4*)(Bs + krow * 136 + nq) = d;
      if (k0 + 32 < K)
        __builtin_prefetch(Bm + (size_t)(k0 + 32 + krow) * N + n0 + nq, 0, 1);
    }
#endif
    __syncthreads();

    FragB16 af[4];
    #pragma unroll
    for (int t = 0; t < 4; ++t) {
      int row = mw * 64 + t * 16 + r;
      #pragma unroll
      for (int j = 0; j < 8; ++j)
        af[t].u[j] = *(const unsigned*)(As + row * 40 + frag_k(j, kh));
    }
    FragB16 bfr[2];
    #pragma unroll
    for (int u2 = 0; u2 < 2; ++u2) {
      int col = nw * 32 + u2 * 16 + r;
      #pragma unroll
      for (int j = 0; j < 8; ++j) {
        int k = frag_k(j, kh);
        unsigned lo = *(const unsigned short*)(Bs + k * 136 + col);
        unsigned hi = *(const unsigned short*)(Bs + (k + 1) * 136 + col);
        bfr[u2].u[j] = lo | (hi << 16);
      }
    }
    #pragma unroll
    for (int t = 0; t < 4; ++t)
      #pragma unroll
      for (int u2 = 0; u2 < 2; ++u2)
        acc[t][u2] = wmma_bf16(af[t], bfr[u2], acc[t][u2]);
  }

  // C layout: vgpr i -> M = i + 8*(lane>=16), N = lane&15
  #pragma unroll
  for (int t = 0; t < 4; ++t)
    #pragma unroll
    for (int u2 = 0; u2 < 2; ++u2)
      #pragma unroll
      for (int i = 0; i < 8; ++i) {
        int row = m0 + mw * 64 + t * 16 + (lane >> 4) * 8 + i;
        int col = n0 + nw * 32 + u2 * 16 + (lane & 15);
        C[(size_t)row * N + col] = acc[t][u2][i];
      }
}

// ------------------------------------------------------------------- RoPE
// dst = (x*cos + rotate_half(x)*sin) * scale  (bf16); optional fp32 present write.
__global__ void rope_apply(const float* __restrict__ src,
                           const float* __restrict__ cosT,
                           const float* __restrict__ sinT,
                           bf16_t* __restrict__ dst,
                           float* __restrict__ present,  // base at slot, or null
                           int nHeads, float scale) {
  int idx = blockIdx.x * blockDim.x + threadIdx.x;
  int total = B_DIM * S_DIM * nHeads * (HD / 2);
  if (idx >= total) return;
  int d = idx & 63;
  int h = (idx >> 6) % nHeads;
  int s = (idx / (64 * nHeads)) % S_DIM;
  int b = idx / (64 * nHeads * S_DIM);
  size_t base = (((size_t)(b * S_DIM + s)) * nHeads + h) * HD;
  float x1 = src[base + d], x2 = src[base + d + 64];
  float c1 = cosT[s * HD + d],  c2 = cosT[s * HD + d + 64];
  float s1 = sinT[s * HD + d],  s2 = sinT[s * HD + d + 64];
  float o1 = x1 * c1 - x2 * s1;
  float o2 = x2 * c2 + x1 * s2;
  dst[base + d]      = (bf16_t)(o1 * scale);
  dst[base + d + 64] = (bf16_t)(o2 * scale);
  if (present) {
    size_t pi = ((size_t)b * (2 * NKVH) + h) * MAXPOS * HD + (size_t)s * HD;
    present[pi + d]      = o1;
    present[pi + d + 64] = o2;
  }
}

__global__ void v_store(const float* __restrict__ src, bf16_t* __restrict__ dst,
                        float* __restrict__ presentV /* base at slot 1 */) {
  int idx = blockIdx.x * blockDim.x + threadIdx.x;
  int total = B_DIM * S_DIM * NKVH * HD;
  if (idx >= total) return;
  int d = idx & 127;
  int h = (idx >> 7) % NKVH;
  int s = (idx / (128 * NKVH)) % S_DIM;
  int b = idx / (128 * NKVH * S_DIM);
  float v = src[idx];
  dst[idx] = (bf16_t)v;
  presentV[((size_t)b * (2 * NKVH) + h) * MAXPOS * HD + (size_t)s * HD + d] = v;
}

// ------------------------------------------------------------ flash attention
// grid: B*NH*(S/128) blocks, 8 waves/block, 1 wave = 16 query rows.
// Q pre-scaled by 1/sqrt(D). Streams 32 keys/iter: 8 WMMA scores + 8 WMMA P@V.
__launch_bounds__(256)
__global__ void flash_attn(const bf16_t* __restrict__ Q,
                           const bf16_t* __restrict__ Kc,
                           const bf16_t* __restrict__ Vc,
                           bf16_t* __restrict__ O) {
  __shared__ bf16_t P[8][16 * 40];   // per-wave 16x32 prob tile, stride 40
  const int tid = threadIdx.x, lane = tid & 31, w = tid >> 5;
  const int qblocks = S_DIM / 128;
  int qb = blockIdx.x % qblocks;
  int h  = (blockIdx.x / qblocks) % NH;
  int b  = blockIdx.x / (qblocks * NH);
  int hk = h / (NH / NKVH);
  int q0 = qb * 128 + w * 16;
  const int r = lane & 15, kh = (lane >> 4) * 8;

  const bf16_t* Qb = Q  + ((size_t)(b * S_DIM) * NH   + h)  * HD;
  const bf16_t* Kb = Kc + ((size_t)(b * S_DIM) * NKVH + hk) * HD;
  const bf16_t* Vb = Vc + ((size_t)(b * S_DIM) * NKVH + hk) * HD;

  // Q A-fragments, 4 k-chunks of 32 over D=128
  FragB16 qf[4];
  {
    size_t qrow = (size_t)(q0 + r) * NH * HD;
    #pragma unroll
    for (int c = 0; c < 4; ++c)
      #pragma unroll
      for (int j = 0; j < 8; ++j)
        qf[c].u[j] = *(const unsigned*)(Qb + qrow + c * 32 + frag_k(j, kh));
  }

  f32x8 o_acc[8] = {};
  float m_run[8], l_run[8];
  #pragma unroll
  for (int i = 0; i < 8; ++i) { m_run[i] = -3.0e38f; l_run[i] = 0.f; }

  const int nb = (q0 + 16 + 31) >> 5;
  for (int kb = 0; kb < nb; ++kb) {
    const int kv0 = kb * 32;

    // scores: Q(16x128) @ K^T(128x32) -> two 16x16 tiles
    f32x8 sc[2] = {};
    #pragma unroll
    for (int kn = 0; kn < 2; ++kn) {
      size_t krow = (size_t)(kv0 + kn * 16 + r) * NKVH * HD;
      #pragma unroll
      for (int c = 0; c < 4; ++c) {
        FragB16 kf;
        #pragma unroll
        for (int j = 0; j < 8; ++j)
          kf.u[j] = *(const unsigned*)(Kb + krow + c * 32 + frag_k(j, kh));
        sc[kn] = wmma_bf16(qf[c], kf, sc[kn]);
      }
    }

    // causal mask + online softmax (row lives in one VGPR across a 16-lane half)
    const int rowBase = q0 + (lane >> 4) * 8;
    #pragma unroll
    for (int i = 0; i < 8; ++i) {
      int qrow = rowBase + i;
      if (kv0 + r > qrow)      sc[0][i] = -3.0e38f;
      if (kv0 + 16 + r > qrow) sc[1][i] = -3.0e38f;
      float mx = fmaxf(sc[0][i], sc[1][i]);
      mx = fmaxf(mx, __shfl_xor(mx, 1, 32));
      mx = fmaxf(mx, __shfl_xor(mx, 2, 32));
      mx = fmaxf(mx, __shfl_xor(mx, 4, 32));
      mx = fmaxf(mx, __shfl_xor(mx, 8, 32));
      float newm  = fmaxf(m_run[i], mx);
      float scale = __expf(m_run[i] - newm);
      float p0 = __expf(sc[0][i] - newm);
      float p1 = __expf(sc[1][i] - newm);
      float rs = p0 + p1;
      rs += __shfl_xor(rs, 1, 32);
      rs += __shfl_xor(rs, 2, 32);
      rs += __shfl_xor(rs, 4, 32);
      rs += __shfl_xor(rs, 8, 32);
      l_run[i] = l_run[i] * scale + rs;
      m_run[i] = newm;
      #pragma unroll
      for (int n = 0; n < 8; ++n) o_acc[n][i] *= scale;
      // C-layout -> LDS (row-major 16x32) for A-layout reload
      int pr = (lane >> 4) * 8 + i;
      P[w][pr * 40 + r]      = (bf16_t)p0;
      P[w][pr * 40 + 16 + r] = (bf16_t)p1;
    }

    asm volatile("s_wait_dscnt 0x0" ::: "memory");   // cross-lane LDS RAW in-wave

    FragB16 pf;
    #pragma unroll
    for (int j = 0; j < 8; ++j)
      pf.u[j] = *(const unsigned*)(&P[w][r * 40 + frag_k(j, kh)]);

    // O += P(16x32) @ V(32x128)
    #pragma unroll
    for (int vd = 0; vd < 8; ++vd) {
      FragB16 vf;
      int dcol = vd * 16 + r;
      #pragma unroll
      for (int j = 0; j < 8; ++j) {
        int kv = kv0 + frag_k(j, kh);
        unsigned lo = *(const unsigned short*)(Vb + (size_t)kv * NKVH * HD + dcol);
        unsigned hi = *(const unsigned short*)(Vb + (size_t)(kv + 1) * NKVH * HD + dcol);
        vf.u[j] = lo | (hi << 16);
      }
      o_acc[vd] = wmma_bf16(pf, vf, o_acc[vd]);
    }
  }

  // epilogue: O / l, write attn[b,s,h,d] as bf16
  #pragma unroll
  for (int i = 0; i < 8; ++i) {
    float inv = 1.0f / l_run[i];
    int qrow = q0 + (lane >> 4) * 8 + i;
    size_t obase = ((size_t)(b * S_DIM + qrow) * NH + h) * HD;
    #pragma unroll
    for (int vd = 0; vd < 8; ++vd)
      O[obase + vd * 16 + (lane & 15)] = (bf16_t)(o_acc[vd][i] * inv);
  }
}

// ------------------------------------------------------------------ launch
extern "C" void kernel_launch(void* const* d_in, const int* in_sizes, int n_in,
                              void* d_out, int out_size, void* d_ws, size_t ws_size,
                              hipStream_t stream) {
  (void)in_sizes; (void)n_in; (void)out_size; (void)ws_size;
  const float* hidden = (const float*)d_in[0];
  const float* past   = (const float*)d_in[1];
  const float* cosT   = (const float*)d_in[2];
  const float* sinT   = (const float*)d_in[3];
  const float* Wq     = (const float*)d_in[4];
  const float* Wk     = (const float*)d_in[5];
  const float* Wv     = (const float*)d_in[6];
  const float* Wo     = (const float*)d_in[7];

  float* out     = (float*)d_out;
  float* present = out + (size_t)B_DIM * S_DIM * HID_;

  const size_t nX  = (size_t)B_DIM * S_DIM * HID_;      // 16.7M
  const size_t nWq = (size_t)HID_ * NH * HD;            // 16.7M
  const size_t nWk = (size_t)HID_ * NKVH * HD;          // 4.2M
  const size_t nQ  = (size_t)B_DIM * S_DIM * NH * HD;   // 16.7M
  const size_t nK  = (size_t)B_DIM * S_DIM * NKVH * HD; // 4.2M
  const size_t nP  = (size_t)B_DIM * 2 * NKVH * MAXPOS * HD;

  char* ws = (char*)d_ws;
  size_t off = 0;
  auto alloc = [&](size_t bytes) -> char* {
    char* p = ws + off; off += (bytes + 255) & ~(size_t)255; return p;
  };
  bf16_t* Xb  = (bf16_t*)alloc(nX  * 2);
  bf16_t* Wqb = (bf16_t*)alloc(nWq * 2);
  bf16_t* Wkb = (bf16_t*)alloc(nWk * 2);
  bf16_t* Wvb = (bf16_t*)alloc(nWk * 2);
  bf16_t* Wob = (bf16_t*)alloc(nWq * 2);
  float*  Qf  = (float*) alloc(nQ * 4);
  float*  Kf  = (float*) alloc(nK * 4);
  float*  Vf  = (float*) alloc(nK * 4);
  bf16_t* Qr  = (bf16_t*)alloc(nQ * 2);
  bf16_t* Kr  = (bf16_t*)alloc(nK * 2);
  bf16_t* Vr  = (bf16_t*)alloc(nK * 2);
  bf16_t* Ab  = (bf16_t*)alloc(nQ * 2);

  const int T = 256;
  // present = past (zero tail included)
  copy_f32x4<<<(int)((nP / 4 + T - 1) / T), T, 0, stream>>>(past, present, (int)(nP / 4));
  // bf16 conversions
  cvt_f32_bf16<<<(int)((nX  / 4 + T - 1) / T), T, 0, stream>>>(hidden, Xb,  (int)(nX  / 4));
  cvt_f32_bf16<<<(int)((nWq / 4 + T - 1) / T), T, 0, stream>>>(Wq,     Wqb, (int)(nWq / 4));
  cvt_f32_bf16<<<(int)((nWk / 4 + T - 1) / T), T, 0, stream>>>(Wk,     Wkb, (int)(nWk / 4));
  cvt_f32_bf16<<<(int)((nWk / 4 + T - 1) / T), T, 0, stream>>>(Wv,     Wvb, (int)(nWk / 4));
  cvt_f32_bf16<<<(int)((nWq / 4 + T - 1) / T), T, 0, stream>>>(Wo,     Wob, (int)(nWq / 4));

  const int M = B_DIM * S_DIM;                // 4096
  // projections
  gemm_bf16_wmma<<<dim3((NH * HD) / 128, M / 128), T, 0, stream>>>(Xb, Wqb, Qf, M, NH * HD, HID_);
  gemm_bf16_wmma<<<dim3((NKVH * HD) / 128, M / 128), T, 0, stream>>>(Xb, Wkb, Kf, M, NKVH * HD, HID_);
  gemm_bf16_wmma<<<dim3((NKVH * HD) / 128, M / 128), T, 0, stream>>>(Xb, Wvb, Vf, M, NKVH * HD, HID_);

  // RoPE (+ present writes); fold 1/sqrt(D) into Q
  const float qscale = 0.08838834764831845f;  // 1/sqrt(128)
  int nRq = B_DIM * S_DIM * NH * (HD / 2);
  int nRk = B_DIM * S_DIM * NKVH * (HD / 2);
  rope_apply<<<(nRq + T - 1) / T, T, 0, stream>>>(Qf, cosT, sinT, Qr, nullptr, NH, qscale);
  rope_apply<<<(nRk + T - 1) / T, T, 0, stream>>>(Kf, cosT, sinT, Kr, present, NKVH, 1.0f);
  int nV = B_DIM * S_DIM * NKVH * HD;
  v_store<<<(nV + T - 1) / T, T, 0, stream>>>(Vf, Vr, present + (size_t)NKVH * MAXPOS * HD);

  // attention
  flash_attn<<<B_DIM * NH * (S_DIM / 128), T, 0, stream>>>(Qr, Kr, Vr, Ab);

  // output projection
  gemm_bf16_wmma<<<dim3(HID_ / 128, M / 128), T, 0, stream>>>(Ab, Wob, out, M, HID_, HID_);
}
